// unit_gcn_58987080843778
// MI455X (gfx1250) — matrix operand order repaired and checked by reading, not compiled, not run on hardware
//
#include <hip/hip_runtime.h>
#include <hip/hip_bf16.h>

// ---------------------------------------------------------------------------
// unit_gcn for MI455X (gfx1250).
//  - All 1x1 convs (86% of FLOPs) via v_wmma_f32_16x16x32_bf16, X tile staged
//    once per block in LDS in WMMA-B-fragment order (ds_load_b128 reads).
//  - scores kernel stages chunks with global_load_async_to_lds_b128 (+
//    s_wait_asynccnt), the CDNA5 async-copy path.
// ---------------------------------------------------------------------------

#define B_    64
#define CIN   64
#define T_    300
#define J_    25
#define TJ    7500        // T_*J_ : N dim per batch for the conv GEMMs
#define INTER 32
#define COUT  128
#define NSTRIP 128        // N columns staged per block (8 subtiles of 16)

typedef __attribute__((ext_vector_type(16))) __bf16 v16bf;
typedef __attribute__((ext_vector_type(8)))  float  v8f;

__device__ __forceinline__ unsigned short f2bf(float f) {
    return __builtin_bit_cast(unsigned short, (__bf16)f);   // native cvt
}
__device__ __forceinline__ float bf2f(unsigned short h) {
    return __builtin_bit_cast(float, ((unsigned)h) << 16);
}

// async 16B global -> LDS copy (ASYNCcnt-tracked); inline asm per ISA 15.18.3
__device__ __forceinline__ void async_g2l_b128(unsigned lds_byte_addr, const void* gaddr) {
    asm volatile("global_load_async_to_lds_b128 %0, %1, off"
                 :: "v"(lds_byte_addr), "v"(gaddr) : "memory");
}
__device__ __forceinline__ void wait_async0() {
    asm volatile("s_wait_asynccnt 0x0" ::: "memory");
}

// ---------------------------------------------------------------------------
// Kernel 0: cast x (f32) -> bf16 once; reused by 7 WMMA conv passes.
// ---------------------------------------------------------------------------
__global__ __launch_bounds__(256)
void cast_bf16_kernel(const float* __restrict__ in, unsigned short* __restrict__ out, int n) {
    int i = blockIdx.x * blockDim.x + threadIdx.x;
    int stride = gridDim.x * blockDim.x;
    for (; i < n; i += stride) out[i] = f2bf(in[i]);
}

// ---------------------------------------------------------------------------
// WMMA 1x1 conv:  Y[b,o,n] = sum_c W[o,c] * X[b,c,n] + bias[o]     (K = 64)
// Block = 256 thr (8 waves) on one (b, 128-wide N strip).
// LDS holds the X tile [64K x 128N] pre-swizzled into B-fragment order:
//   element (k,n): g=k>>5, half=(k>>4)&1, e=k&15, s=nl>>4, lane=(half<<4)|(nl&15)
//   offset = ((g*8+s)<<9) + (lane<<4) + e     -> fragment read = 32B contiguous
// Wave w: M-tile = w%MT, subtiles s = (w/MT)*MT .. +MT-1; 2 WMMA per subtile.
// MODE 0: bf16 out (E1/E2)   MODE 1: f32 '='/'+='   MODE 2: fused BN+res+relu
// ---------------------------------------------------------------------------
template <int MODE, int MT>
__global__ __launch_bounds__(256)
void conv1x1_wmma_kernel(const float* __restrict__ W, const float* __restrict__ bias,
                         const unsigned short* __restrict__ X, void* __restrict__ Out,
                         int accumulate, const float* __restrict__ Yin,
                         const float* __restrict__ bng, const float* __restrict__ bnb,
                         const float* __restrict__ bnm, const float* __restrict__ bnv,
                         const float* __restrict__ dg,  const float* __restrict__ db,
                         const float* __restrict__ dm,  const float* __restrict__ dv) {
    constexpr int O = MT * 16;
    __shared__ __align__(32) unsigned short xs[64 * NSTRIP];   // 16 KB

    const int tid = threadIdx.x;
    const int b   = blockIdx.y;
    const int n0  = blockIdx.x * NSTRIP;
    const unsigned short* Xb = X + (size_t)b * CIN * TJ;

    // ---- cooperative swizzled fill: 1024 chunks of 8 bf16 -----------------
    const bool edge = (n0 + NSTRIP > TJ);
#pragma unroll
    for (int it = 0; it < 4; ++it) {
        const int ch  = tid + it * 256;        // 0..1023
        const int k   = ch >> 4;               // row (input channel)
        const int ncl = (ch & 15) << 3;        // col within strip (mult of 8)
        const int g = k >> 5, half = (k >> 4) & 1, e = k & 15;
        const int base_e = (((g << 3) + (ncl >> 4)) << 9)
                         + (((half << 4) | (ncl & 15)) << 4) + e;
        if (!edge) {
            const unsigned short* gp = Xb + (size_t)k * TJ + n0 + ncl;   // 8B-aligned pair of u64 loads
            const unsigned long long q0 = *reinterpret_cast<const unsigned long long*>(gp);
            const unsigned long long q1 = *reinterpret_cast<const unsigned long long*>(gp + 4);
#pragma unroll
            for (int i = 0; i < 4; ++i) xs[base_e + (i << 4)]       = (unsigned short)(q0 >> (16 * i));
#pragma unroll
            for (int i = 0; i < 4; ++i) xs[base_e + ((i + 4) << 4)] = (unsigned short)(q1 >> (16 * i));
        } else {
#pragma unroll
            for (int i = 0; i < 8; ++i) {
                int n = n0 + ncl + i; n = n < TJ ? n : TJ - 1;   // clamp; masked at store
                xs[base_e + (i << 4)] = Xb[(size_t)k * TJ + n];
            }
        }
    }
    __syncthreads();

    // ---- per-wave WMMA ----------------------------------------------------
    const int w    = tid >> 5;
    const int lane = tid & 31;
    const int half = lane >> 4;
    const int mr   = lane & 15;
    const int mt   = w % MT;
    const int sg   = w / MT;
    const int o0   = mt << 4;

    // A operand: W tile [16M x 64K] -> two 16x32 bf16 fragments (k = 16*half + e, same order as B)
    const int wrow = (o0 + mr) * CIN;
    v16bf a0{}, a1{};
#pragma unroll
    for (int e = 0; e < 16; ++e) {
        const int koff = (half << 4) + e;
        a0[e] = (__bf16)W[wrow + koff];
        a1[e] = (__bf16)W[wrow + 32 + koff];
    }

    float bias8[8], ysc[8], ysh[8], rsc[8], rsh[8];
#pragma unroll
    for (int r = 0; r < 8; ++r) {
        const int o = o0 + (half << 3) + r;
        bias8[r] = bias[o];
        if (MODE == 2) {
            const float s1 = bng[o] * rsqrtf(bnv[o] + 1e-5f);
            ysc[r] = s1; ysh[r] = bnb[o] - bnm[o] * s1;
            const float s2 = dg[o] * rsqrtf(dv[o] + 1e-5f);
            rsc[r] = s2; rsh[r] = db[o] - dm[o] * s2;
        }
    }

#pragma unroll
    for (int j = 0; j < MT; ++j) {
        const int s    = sg * MT + j;                 // subtile 0..7
        const int ncol = n0 + (s << 4) + mr;

        // B fragments: 32B contiguous LDS reads (2x ds_load_b128 each)
        const v16bf b0 = *reinterpret_cast<const v16bf*>(&xs[(s << 9) + (lane << 4)]);
        const v16bf b1 = *reinterpret_cast<const v16bf*>(&xs[((8 + s) << 9) + (lane << 4)]);

        v8f c{};
        c = __builtin_amdgcn_wmma_f32_16x16x32_bf16(false, a0, false, b0,
                                                    (short)0, c, false, false);
        c = __builtin_amdgcn_wmma_f32_16x16x32_bf16(false, a1, false, b1,
                                                    (short)0, c, false, false);

        if (ncol < TJ) {   // predicated epilogue only; WMMA ran with EXEC all-1s
#pragma unroll
            for (int r = 0; r < 8; ++r) {
                const int o = o0 + (half << 3) + r;   // D: m = half*8 + r, n = mr
                const size_t oidx = ((size_t)b * O + o) * TJ + ncol;
                const float v = c[r] + bias8[r];
                if (MODE == 0) {
                    ((unsigned short*)Out)[oidx] = f2bf(v);
                } else if (MODE == 1) {
                    float* of = (float*)Out;
                    of[oidx] = accumulate ? (of[oidx] + v) : v;
                } else {
                    const float y = Yin[oidx];
                    const float t = y * ysc[r] + ysh[r] + v * rsc[r] + rsh[r];
                    ((float*)Out)[oidx] = t > 0.0f ? t : 0.0f;
                }
            }
        }
    }
}

// ---------------------------------------------------------------------------
// scores[b,j1,j2] = (1/9600) * sum_{c<32,t} E1[b,c,t,j1]*E2[b,c,t,j2]
// One block per batch. 1 KB contiguous chunks staged with
// global_load_async_to_lds_b128, drained by s_wait_asynccnt, converted once
// to f32 in LDS, then pure-FMA accumulation.
// ---------------------------------------------------------------------------
__global__ __launch_bounds__(256)
void scores_kernel(const unsigned short* __restrict__ E1,
                   const unsigned short* __restrict__ E2,
                   float* __restrict__ scores) {
    __shared__ __align__(16) unsigned short u1[512], u2[512];
    __shared__ float s1[500], s2[500];
    const int b = blockIdx.x, tid = threadIdx.x;
    float acc0 = 0.f, acc1 = 0.f, acc2 = 0.f;
    const size_t base = (size_t)b * INTER * TJ;

    for (int c = 0; c < INTER; ++c) {
        for (int tc = 0; tc < 15; ++tc) {                 // 15 * 20t = 300 = T
            const size_t off = base + (size_t)c * TJ + (size_t)tc * 500;
            // stage 504 bf16 per tensor (63 x 16B async transfers each)
            if (tid < 63) {
                async_g2l_b128((unsigned)(size_t)&u1[tid << 3], (const void*)(E1 + off + (tid << 3)));
            } else if (tid >= 64 && tid < 127) {
                const int q = tid - 64;
                async_g2l_b128((unsigned)(size_t)&u2[q << 3], (const void*)(E2 + off + (q << 3)));
            }
            wait_async0();
            __syncthreads();
            for (int k = tid; k < 500; k += 256) { s1[k] = bf2f(u1[k]); s2[k] = bf2f(u2[k]); }
            __syncthreads();
#pragma unroll
            for (int p = 0; p < 3; ++p) {
                const int pair = tid + (p << 8);
                if (pair < 625) {
                    const int j1 = pair / 25, j2 = pair - j1 * 25;
                    float a = 0.f;
#pragma unroll
                    for (int kk = 0; kk < 20; ++kk)
                        a += s1[kk * 25 + j1] * s2[kk * 25 + j2];
                    if (p == 0) acc0 += a; else if (p == 1) acc1 += a; else acc2 += a;
                }
            }
            __syncthreads();   // protect s1/s2 (and u1/u2) before next chunk
        }
    }
    const float inv = 1.0f / 9600.0f;
#pragma unroll
    for (int p = 0; p < 3; ++p) {
        const int pair = tid + (p << 8);
        if (pair < 625)
            scores[b * 625 + pair] = (p == 0 ? acc0 : (p == 1 ? acc1 : acc2)) * inv;
    }
}

// ---------------------------------------------------------------------------
// Column softmax over j1 (axis -2), then + PA[i] + I. In-place on scores.
// ---------------------------------------------------------------------------
__global__ __launch_bounds__(32)
void softmax_adj_kernel(float* __restrict__ scores, const float* __restrict__ PAi) {
    const int b = blockIdx.x, j2 = threadIdx.x;
    if (j2 >= J_) return;
    float* col = scores + b * 625;
    float mx = -3.4e38f;
    for (int j1 = 0; j1 < J_; ++j1) { const float v = col[j1 * J_ + j2]; mx = v > mx ? v : mx; }
    float e[J_], sum = 0.f;
    for (int j1 = 0; j1 < J_; ++j1) { e[j1] = __expf(col[j1 * J_ + j2] - mx); sum += e[j1]; }
    const float inv = 1.0f / sum;
    for (int j1 = 0; j1 < J_; ++j1)
        col[j1 * J_ + j2] = e[j1] * inv + PAi[j1 * J_ + j2] + (j1 == j2 ? 1.0f : 0.0f);
}

// ---------------------------------------------------------------------------
// z[b,c,t,j2] = sum_j1 x[b,c,t,j1] * Aadapt[b,j1,j2]  -> bf16 (feeds Wd WMMA)
// ---------------------------------------------------------------------------
__global__ __launch_bounds__(256)
void aggregate_kernel(const float* __restrict__ x, const float* __restrict__ Aad,
                      unsigned short* __restrict__ zb) {
    const int b = blockIdx.y;
    const int rid = blockIdx.x * 256 + threadIdx.x;      // 75*256 == 19200 == C*T
    __shared__ float As[625];
    for (int k = threadIdx.x; k < 625; k += 256) As[k] = Aad[b * 625 + k];
    __syncthreads();
    const size_t off = (size_t)b * (CIN * TJ) + (size_t)rid * J_;
    float xr[J_];
#pragma unroll
    for (int j = 0; j < J_; ++j) xr[j] = x[off + j];
#pragma unroll
    for (int j2 = 0; j2 < J_; ++j2) {
        float a = 0.f;
#pragma unroll
        for (int j1 = 0; j1 < J_; ++j1) a += xr[j1] * As[j1 * J_ + j2];
        zb[off + j2] = f2bf(a);
    }
}

// ---------------------------------------------------------------------------
extern "C" void kernel_launch(void* const* d_in, const int* in_sizes, int n_in,
                              void* d_out, int out_size, void* d_ws, size_t ws_size,
                              hipStream_t stream) {
    (void)in_sizes; (void)n_in; (void)out_size; (void)ws_size;
    const float* x     = (const float*)d_in[0];
    const float* PA    = (const float*)d_in[1];
    const float* Wa    = (const float*)d_in[2];
    const float* ba    = (const float*)d_in[3];
    const float* Wb    = (const float*)d_in[4];
    const float* bb    = (const float*)d_in[5];
    const float* Wd    = (const float*)d_in[6];
    const float* bd    = (const float*)d_in[7];
    const float* Wdown = (const float*)d_in[8];
    const float* bdown = (const float*)d_in[9];
    const float* bng   = (const float*)d_in[10];
    const float* bnb   = (const float*)d_in[11];
    const float* bnm   = (const float*)d_in[12];
    const float* bnv   = (const float*)d_in[13];
    const float* dg    = (const float*)d_in[14];
    const float* db    = (const float*)d_in[15];
    const float* dm    = (const float*)d_in[16];
    const float* dv    = (const float*)d_in[17];
    float* out = (float*)d_out;

    // workspace carve (offsets 256B aligned; ~184.5 MB total)
    char* ws = (char*)d_ws;
    unsigned short* xb = (unsigned short*)(ws);                    // 61,440,000 B
    unsigned short* E1 = (unsigned short*)(ws + 61440000);         // 30,720,000 B
    unsigned short* E2 = (unsigned short*)(ws + 92160000);         // 30,720,000 B
    unsigned short* zb = (unsigned short*)(ws + 122880000);        // 61,440,000 B
    float*       score = (float*)      (ws + 184320000);           //    160,000 B

    cast_bf16_kernel<<<15000, 256, 0, stream>>>(x, xb, B_ * CIN * TJ);

    const dim3 gC((TJ + NSTRIP - 1) / NSTRIP, B_);                 // 59 x 64
    for (int i = 0; i < 3; ++i) {
        conv1x1_wmma_kernel<0, 2><<<gC, 256, 0, stream>>>(
            Wa + i * INTER * CIN, ba + i * INTER, xb, E1, 0,
            nullptr, nullptr, nullptr, nullptr, nullptr, nullptr, nullptr, nullptr, nullptr);
        conv1x1_wmma_kernel<0, 2><<<gC, 256, 0, stream>>>(
            Wb + i * INTER * CIN, bb + i * INTER, xb, E2, 0,
            nullptr, nullptr, nullptr, nullptr, nullptr, nullptr, nullptr, nullptr, nullptr);
        scores_kernel<<<B_, 256, 0, stream>>>(E1, E2, score);
        softmax_adj_kernel<<<B_, 32, 0, stream>>>(score, PA + i * J_ * J_);
        aggregate_kernel<<<dim3(75, B_), 256, 0, stream>>>(x, score, zb);
        conv1x1_wmma_kernel<1, 8><<<gC, 256, 0, stream>>>(
            Wd + i * COUT * CIN, bd + i * COUT, zb, out, i > 0 ? 1 : 0,
            nullptr, nullptr, nullptr, nullptr, nullptr, nullptr, nullptr, nullptr, nullptr);
    }
    // fused: res = conv(Wdown, x); out = relu(bn(y) + bn_down(res)) in place
    conv1x1_wmma_kernel<2, 8><<<gC, 256, 0, stream>>>(
        Wdown, bdown, xb, out, 0, out, bng, bnb, bnm, bnv, dg, db, dm, dv);
}
